// MoE_13846974562945
// MI455X (gfx1250) — matrix operand order repaired and checked by reading, not compiled, not run on hardware
//
#include <hip/hip_runtime.h>
#include <math.h>

// ---------------------------------------------------------------------------
// MoE (D=1024, H=4096, E=8, top-2) for MI455X / gfx1250
// bf16 WMMA grouped GEMM + async global->LDS double buffering (when available)
// ---------------------------------------------------------------------------

typedef __attribute__((ext_vector_type(8)))  __bf16 v8bf;
typedef __attribute__((ext_vector_type(16))) __bf16 v16bf;
typedef __attribute__((ext_vector_type(8)))  float  v8f;
typedef __attribute__((ext_vector_type(4)))  int    v4i;

#define DEVINL __device__ __forceinline__

constexpr int D_ = 1024;   // model dim
constexpr int H_ = 4096;   // hidden dim
constexpr int E_ = 8;      // experts
constexpr int T_ = 16384;  // tokens (4 * 4096)

constexpr int BM  = 128;   // block tile M
constexpr int BN  = 128;   // block tile N
constexpr int BK  = 32;    // K step (one wmma K)
constexpr int LDR = 40;    // LDS row stride in bf16 elems (32 + 8 pad)

// workspace layout (bytes)
constexpr size_t O_CNT = 0;                                  // E ints
constexpr size_t O_OFF = 256;                                // E+1 ints
constexpr size_t O_TOK = 512;                                // E*T ints
constexpr size_t O_WGT = O_TOK + (size_t)E_ * T_ * 4;        // E*T floats
constexpr size_t O_W1T = O_WGT + (size_t)E_ * T_ * 4;        // E*H*D bf16
constexpr size_t O_W2T = O_W1T + (size_t)E_ * H_ * D_ * 2;   // E*D*H bf16
constexpr size_t O_HB  = O_W2T + (size_t)E_ * D_ * H_ * 2;   // hidden rows bf16

DEVINL unsigned short f2bf(float f) {                  // round-to-nearest-even
  unsigned u = __float_as_uint(f);
  return (unsigned short)((u + 0x7FFFu + ((u >> 16) & 1u)) >> 16);
}
DEVINL unsigned pk2(unsigned short a, unsigned short b) {
  return (unsigned)a | ((unsigned)b << 16);
}
#define SHUF16(lo, hi) \
  __builtin_shufflevector(lo, hi, 0,1,2,3,4,5,6,7,8,9,10,11,12,13,14,15)

// ------------------- CDNA5 async global->LDS (guarded) ---------------------
#if __has_builtin(__builtin_amdgcn_global_load_async_to_lds_b128)
#define HAS_ASYNC_LDS 1
#else
#define HAS_ASYNC_LDS 0
#endif

typedef __attribute__((address_space(1))) v4i* gptr4;   // global int4*
typedef __attribute__((address_space(3))) v4i* lptr4;   // LDS int4*

DEVINL void async_cp16(const void* g, void* l) {
#if HAS_ASYNC_LDS
  __builtin_amdgcn_global_load_async_to_lds_b128((gptr4)g, (lptr4)l, 0, 0);
#else
  *reinterpret_cast<uint4*>(l) = *reinterpret_cast<const uint4*>(g);
#endif
}

DEVINL void wait_async0() {
#if HAS_ASYNC_LDS
#if __has_builtin(__builtin_amdgcn_s_wait_asynccnt)
  __builtin_amdgcn_s_wait_asynccnt(0);
#else
  asm volatile("s_wait_asynccnt 0x0" ::: "memory");
#endif
#endif
}

// --------------------------- weight cast + transpose -----------------------
// src: [E][R][C] fp32 row-major  ->  dst: [E][C][R] bf16 row-major
__global__ __launch_bounds__(256) void wcast_tr_kernel(
    const float* __restrict__ src, unsigned short* __restrict__ dst,
    int R, int C) {
  __shared__ unsigned short tile[32][33];
  const int e  = blockIdx.z;
  const int c0 = blockIdx.x * 32, r0 = blockIdx.y * 32;
  const int tx = threadIdx.x, ty = threadIdx.y;          // block (32,8)
  const float* s = src + (size_t)e * R * C;
  unsigned short* d = dst + (size_t)e * (size_t)R * C;
#pragma unroll
  for (int i = 0; i < 4; ++i)
    tile[ty + 8 * i][tx] = f2bf(s[(size_t)(r0 + ty + 8 * i) * C + c0 + tx]);
  __syncthreads();
#pragma unroll
  for (int i = 0; i < 4; ++i)
    d[(size_t)(c0 + ty + 8 * i) * R + r0 + tx] = tile[tx][ty + 8 * i];
}

// --------------------------------- gating ----------------------------------
__global__ __launch_bounds__(256) void gate_kernel(
    const float* __restrict__ x, const float* __restrict__ gw,
    const float* __restrict__ gb, int* __restrict__ cnt,
    int* __restrict__ tokList, float* __restrict__ wList) {
  const int wid  = threadIdx.x >> 5;
  const int lane = threadIdx.x & 31;
  const int token = blockIdx.x * 8 + wid;
  const float* xr = x + (size_t)token * D_;
  float acc[E_] = {0.f, 0.f, 0.f, 0.f, 0.f, 0.f, 0.f, 0.f};
  for (int d = lane; d < D_; d += 32) {
    const float xv = xr[d];
    const float* g = gw + d * E_;
#pragma unroll
    for (int e = 0; e < E_; ++e) acc[e] += xv * g[e];
  }
#pragma unroll
  for (int e = 0; e < E_; ++e) {
#pragma unroll
    for (int off = 16; off > 0; off >>= 1)
      acc[e] += __shfl_xor(acc[e], off, 32);
    acc[e] += gb[e];
  }
  if (lane == 0) {
    int e1 = 0; float l1 = acc[0];
#pragma unroll
    for (int e = 1; e < E_; ++e)
      if (acc[e] > l1) { l1 = acc[e]; e1 = e; }
    int e2 = (e1 == 0) ? 1 : 0; float l2 = acc[e2];
#pragma unroll
    for (int e = 0; e < E_; ++e)
      if (e != e1 && acc[e] > l2) { l2 = acc[e]; e2 = e; }
    const float z  = __expf(l2 - l1);          // <= 1
    const float w1 = 1.f / (1.f + z);
    const float w2 = z / (1.f + z);
    int s1 = atomicAdd(&cnt[e1], 1);
    tokList[e1 * T_ + s1] = token; wList[e1 * T_ + s1] = w1;
    int s2 = atomicAdd(&cnt[e2], 1);
    tokList[e2 * T_ + s2] = token; wList[e2 * T_ + s2] = w2;
  }
}

// ------------------------------ offsets scan -------------------------------
__global__ void scan_kernel(const int* __restrict__ cnt, int* __restrict__ offs) {
  if (threadIdx.x == 0 && blockIdx.x == 0) {
    int o = 0;
    for (int e = 0; e < E_; ++e) {
      offs[e] = o;
      o += (cnt[e] + BM - 1) & ~(BM - 1);
    }
    offs[E_] = o;
  }
}

// ------------------------------ GEMM 1 -------------------------------------
// hb[slot, :] = relu( x[tok[slot], :] @ W1[e] + b1[e] )  (bf16 out)
// B tile uses async global->LDS DMA overlapped with the fp32 x gather+convert.
__global__ __launch_bounds__(256) void moe_gemm1(
    const float* __restrict__ x, const unsigned short* __restrict__ w1t,
    const float* __restrict__ b1, const int* __restrict__ cnt,
    const int* __restrict__ offs, const int* __restrict__ tokList,
    unsigned short* __restrict__ hb) {
  const int e     = blockIdx.z;
  const int cnt_e = cnt[e];
  const int mtile = blockIdx.y;
  if (mtile * BM >= cnt_e) return;
  const int ntile = blockIdx.x;
  const int obase = offs[e];

  __shared__ unsigned short lsA[BM * LDR];
  __shared__ unsigned short lsB[BN * LDR];

  const int tid = threadIdx.x, lane = tid & 31, wid = tid >> 5;
  const int l15 = lane & 15, kh = lane >> 4;
  const int M0 = (wid & 1) * 64, N0 = (wid >> 1) * 32;

  const v8f zero = {0.f, 0.f, 0.f, 0.f, 0.f, 0.f, 0.f, 0.f};
  v8f acc[4][2];
#pragma unroll
  for (int mi = 0; mi < 4; ++mi)
#pragma unroll
    for (int ni = 0; ni < 2; ++ni) acc[mi][ni] = zero;

  const unsigned short* bsrc =
      w1t + (size_t)e * H_ * D_ + (size_t)(ntile * BN) * D_;
  const int* tlist = tokList + e * T_;

  for (int kk = 0; kk < D_; kk += BK) {
    __syncthreads();   // previous compute finished reading LDS
    // B tile: async DMA of bf16 W1^T rows (k-contiguous) into LDS
    for (int idx = tid; idx < BN * 4; idx += 256) {
      const int rN = idx >> 2, kc = (idx & 3) * 8;
      async_cp16(bsrc + (size_t)rN * D_ + kk + kc, &lsB[rN * LDR + kc]);
    }
    // A tile meanwhile: gather fp32 rows of x, convert to bf16 (VALU work)
    for (int idx = tid; idx < BM * 4; idx += 256) {
      const int r = idx >> 2, kc = (idx & 3) * 8;
      const int sr = mtile * BM + r;
      uint4 v = {0u, 0u, 0u, 0u};
      if (sr < cnt_e) {
        const float4* p = reinterpret_cast<const float4*>(
            x + (size_t)tlist[sr] * D_ + kk + kc);
        const float4 f0 = p[0], f1 = p[1];
        v.x = pk2(f2bf(f0.x), f2bf(f0.y));
        v.y = pk2(f2bf(f0.z), f2bf(f0.w));
        v.z = pk2(f2bf(f1.x), f2bf(f1.y));
        v.w = pk2(f2bf(f1.z), f2bf(f1.w));
      }
      *reinterpret_cast<uint4*>(&lsA[r * LDR + kc]) = v;
    }
    wait_async0();
    __syncthreads();

    v16bf af[4], bfv[2];
#pragma unroll
    for (int mi = 0; mi < 4; ++mi) {
      const int m = M0 + mi * 16 + l15;
      const v8bf lo = *reinterpret_cast<const v8bf*>(&lsA[m * LDR + kh * 8]);
      const v8bf hi = *reinterpret_cast<const v8bf*>(&lsA[m * LDR + 16 + kh * 8]);
      af[mi] = SHUF16(lo, hi);
    }
#pragma unroll
    for (int ni = 0; ni < 2; ++ni) {
      const int n = N0 + ni * 16 + l15;
      const v8bf lo = *reinterpret_cast<const v8bf*>(&lsB[n * LDR + kh * 8]);
      const v8bf hi = *reinterpret_cast<const v8bf*>(&lsB[n * LDR + 16 + kh * 8]);
      bfv[ni] = SHUF16(lo, hi);
    }
#pragma unroll
    for (int mi = 0; mi < 4; ++mi)
#pragma unroll
      for (int ni = 0; ni < 2; ++ni)
        acc[mi][ni] = __builtin_amdgcn_wmma_f32_16x16x32_bf16(
            false, af[mi], false, bfv[ni], (short)0, acc[mi][ni], false, false);
  }

  // epilogue: +b1, relu, bf16 store
#pragma unroll
  for (int mi = 0; mi < 4; ++mi) {
#pragma unroll
    for (int r = 0; r < 8; ++r) {
      const int sr = mtile * BM + M0 + mi * 16 + kh * 8 + r;
      if (sr < cnt_e) {
#pragma unroll
        for (int ni = 0; ni < 2; ++ni) {
          const int col = ntile * BN + N0 + ni * 16 + l15;
          float v = acc[mi][ni][r] + b1[e * H_ + col];
          v = fmaxf(v, 0.0f);
          hb[(size_t)(obase + sr) * H_ + col] = f2bf(v);
        }
      }
    }
  }
}

// ------------------------------ GEMM 2 -------------------------------------
// out[tok[slot], :] += w[slot] * ( hb[slot, :] @ W2[e] + b2[e] )
// Both tiles are bf16 contiguous copies -> async DMA, double buffered.
__global__ __launch_bounds__(256) void moe_gemm2(
    const unsigned short* __restrict__ hb, const unsigned short* __restrict__ w2t,
    const float* __restrict__ b2, const int* __restrict__ cnt,
    const int* __restrict__ offs, const int* __restrict__ tokList,
    const float* __restrict__ wList, float* __restrict__ out) {
  const int e     = blockIdx.z;
  const int cnt_e = cnt[e];
  const int mtile = blockIdx.y;
  if (mtile * BM >= cnt_e) return;
  const int ntile = blockIdx.x;
  const int obase = offs[e];

  __shared__ unsigned short lsA[2][BM * LDR];
  __shared__ unsigned short lsB[2][BN * LDR];

  const int tid = threadIdx.x, lane = tid & 31, wid = tid >> 5;
  const int l15 = lane & 15, kh = lane >> 4;
  const int M0 = (wid & 1) * 64, N0 = (wid >> 1) * 32;

  const v8f zero = {0.f, 0.f, 0.f, 0.f, 0.f, 0.f, 0.f, 0.f};
  v8f acc[4][2];
#pragma unroll
  for (int mi = 0; mi < 4; ++mi)
#pragma unroll
    for (int ni = 0; ni < 2; ++ni) acc[mi][ni] = zero;

  const unsigned short* asrc = hb + (size_t)(obase + mtile * BM) * H_;
  const unsigned short* bsrc =
      w2t + (size_t)e * D_ * H_ + (size_t)(ntile * BN) * H_;

  auto issue = [&](int kk, int bsel) {
    for (int idx = tid; idx < BM * 4; idx += 256) {
      const int r = idx >> 2, kc = (idx & 3) * 8;
      async_cp16(asrc + (size_t)r * H_ + kk + kc, &lsA[bsel][r * LDR + kc]);
    }
    for (int idx = tid; idx < BN * 4; idx += 256) {
      const int rN = idx >> 2, kc = (idx & 3) * 8;
      async_cp16(bsrc + (size_t)rN * H_ + kk + kc, &lsB[bsel][rN * LDR + kc]);
    }
  };

  constexpr int NK = H_ / BK;   // 128 K-steps
  issue(0, 0);                  // prologue: fill buffer 0

  for (int ks = 0; ks < NK; ++ks) {
    const int cur = ks & 1;
    wait_async0();              // this wave's DMA for step ks complete
    __syncthreads();            // everyone's DMA done + prev compute done
    if (ks + 1 < NK) issue((ks + 1) * BK, cur ^ 1);  // prefetch next step

    v16bf af[4], bfv[2];
#pragma unroll
    for (int mi = 0; mi < 4; ++mi) {
      const int m = M0 + mi * 16 + l15;
      const v8bf lo = *reinterpret_cast<const v8bf*>(&lsA[cur][m * LDR + kh * 8]);
      const v8bf hi = *reinterpret_cast<const v8bf*>(&lsA[cur][m * LDR + 16 + kh * 8]);
      af[mi] = SHUF16(lo, hi);
    }
#pragma unroll
    for (int ni = 0; ni < 2; ++ni) {
      const int n = N0 + ni * 16 + l15;
      const v8bf lo = *reinterpret_cast<const v8bf*>(&lsB[cur][n * LDR + kh * 8]);
      const v8bf hi = *reinterpret_cast<const v8bf*>(&lsB[cur][n * LDR + 16 + kh * 8]);
      bfv[ni] = SHUF16(lo, hi);
    }
#pragma unroll
    for (int mi = 0; mi < 4; ++mi)
#pragma unroll
      for (int ni = 0; ni < 2; ++ni)
        acc[mi][ni] = __builtin_amdgcn_wmma_f32_16x16x32_bf16(
            false, af[mi], false, bfv[ni], (short)0, acc[mi][ni], false, false);
  }

  // epilogue: +b2, scale by gate weight, scatter-add into out
#pragma unroll
  for (int mi = 0; mi < 4; ++mi) {
#pragma unroll
    for (int r = 0; r < 8; ++r) {
      const int sr = mtile * BM + M0 + mi * 16 + kh * 8 + r;
      if (sr < cnt_e) {
        const int   tok = tokList[e * T_ + sr];
        const float wg  = wList[e * T_ + sr];
#pragma unroll
        for (int ni = 0; ni < 2; ++ni) {
          const int col = ntile * BN + N0 + ni * 16 + l15;
          const float v = acc[mi][ni][r] + b2[e * D_ + col];
          atomicAdd(&out[(size_t)tok * D_ + col], wg * v);
        }
      }
    }
  }
}

// ------------------------------- launcher ----------------------------------
extern "C" void kernel_launch(void* const* d_in, const int* in_sizes, int n_in,
                              void* d_out, int out_size, void* d_ws,
                              size_t ws_size, hipStream_t stream) {
  const float* x      = (const float*)d_in[0];
  const float* gate_w = (const float*)d_in[1];
  const float* gate_b = (const float*)d_in[2];
  const float* W1     = (const float*)d_in[3];
  const float* b1     = (const float*)d_in[4];
  const float* W2     = (const float*)d_in[5];
  const float* b2     = (const float*)d_in[6];
  float* out = (float*)d_out;

  char* ws = (char*)d_ws;
  int*            cnt     = (int*)(ws + O_CNT);
  int*            offs    = (int*)(ws + O_OFF);
  int*            tokList = (int*)(ws + O_TOK);
  float*          wList   = (float*)(ws + O_WGT);
  unsigned short* w1t     = (unsigned short*)(ws + O_W1T);
  unsigned short* w2t     = (unsigned short*)(ws + O_W2T);
  unsigned short* hb      = (unsigned short*)(ws + O_HB);

  (void)hipMemsetAsync(cnt, 0, 256, stream);
  (void)hipMemsetAsync(out, 0, (size_t)out_size * sizeof(float), stream);

  // W1: [E][D][H] -> bf16 [E][H][D];  W2: [E][H][D] -> bf16 [E][D][H]
  wcast_tr_kernel<<<dim3(H_ / 32, D_ / 32, E_), dim3(32, 8), 0, stream>>>(
      W1, w1t, D_, H_);
  wcast_tr_kernel<<<dim3(D_ / 32, H_ / 32, E_), dim3(32, 8), 0, stream>>>(
      W2, w2t, H_, D_);

  gate_kernel<<<T_ / 8, 256, 0, stream>>>(x, gate_w, gate_b, cnt, tokList,
                                          wList);
  scan_kernel<<<1, 32, 0, stream>>>(cnt, offs);

  moe_gemm1<<<dim3(H_ / BN, T_ / BM, E_), 256, 0, stream>>>(
      x, w1t, b1, cnt, offs, tokList, hb);
  moe_gemm2<<<dim3(D_ / BN, T_ / BM, E_), 256, 0, stream>>>(
      hb, w2t, b2, cnt, offs, tokList, wList, out);
}
